// MSA_65438121722502
// MI455X (gfx1250) — matrix operand order repaired and checked by reading, not compiled, not run on hardware
//
#include <hip/hip_runtime.h>
#include <hip/hip_bf16.h>

typedef __attribute__((ext_vector_type(8)))  __bf16 v8bf;
typedef __attribute__((ext_vector_type(16))) __bf16 v16bf;
typedef __attribute__((ext_vector_type(8)))  float  v8f;
typedef unsigned int v4u __attribute__((ext_vector_type(4)));
typedef int v8i __attribute__((ext_vector_type(8)));
typedef int v4i __attribute__((ext_vector_type(4)));

#define HW 65536L
#define DIM 192
#define HEADS 4
#define CH 48

union F16u { float4 q[4]; float f[16]; };

// ---------------------------------------------------------------------------
// K0: zero the atomic accumulators (sumsq q/k + Gram) each launch
// ---------------------------------------------------------------------------
__global__ void msa_zero_kernel(float* p, int n) {
    int i = blockIdx.x * 256 + threadIdx.x;
    if (i < n) p[i] = 0.f;
}

// ---------------------------------------------------------------------------
// TDM: DMA a 2D f32 tile (rows x cols, row stride lda_el elements) into LDS.
// D# packing per CDNA5 ISA (2D: groups 2/3 zero).
// This toolchain exposes the 6-arg builtin:
//   (uint32x4 g0, int32x8 g1, int32x4 g2, int32x4 g3, int32x8, i32 cpol)
// ---------------------------------------------------------------------------
#if __has_builtin(__builtin_amdgcn_tensor_load_to_lds)
#define MSA_HAVE_TDM 1
__device__ __forceinline__ void msa_tdm_load_2d(const float* gsrc, unsigned lds_off,
                                                int rows, int cols, int lda_el) {
    const unsigned long long ga = (unsigned long long)(uintptr_t)gsrc;
    v4u g0;
    g0[0] = 1u;                                     // count=1, user descriptor
    g0[1] = lds_off;                                // lds_addr (bytes)
    g0[2] = (unsigned)(ga & 0xffffffffu);           // global_addr[31:0]
    g0[3] = (unsigned)((ga >> 32) & 0x01ffffffu)    // global_addr[56:32]
          | (2u << 30);                             // type = 2 (image)
    v8i g1;
    g1[0] = (int)(2u << 16);                        // data_size = 2 -> 4 bytes
    g1[1] = (int)(((unsigned)cols & 0xffffu) << 16);            // tensor_dim0[15:0]
    g1[2] = (int)((((unsigned)cols >> 16) & 0xffffu)            // tensor_dim0[31:16]
          | (((unsigned)rows & 0xffffu) << 16));                // tensor_dim1[15:0]
    g1[3] = (int)((((unsigned)rows >> 16) & 0xffffu)            // tensor_dim1[31:16]
          | (((unsigned)cols & 0xffffu) << 16));                // tile_dim0 = cols
    g1[4] = (int)((unsigned)rows & 0xffffu);                    // tile_dim1 = rows, tile_dim2=0
    g1[5] = (int)(unsigned)lda_el;                              // tensor_dim0_stride[31:0]
    g1[6] = 0;                                                  // stride0[47:32], stride1[15:0]
    g1[7] = 0;                                                  // stride1[47:16]
    const v4i z4 = {};
    const v8i z8 = {};
    __builtin_amdgcn_tensor_load_to_lds(g0, g1, z4, z4, z8, 0);
}
#else
#define MSA_HAVE_TDM 0
#endif

// ---------------------------------------------------------------------------
// K1/K5: bf16 WMMA GEMM  C[b] = A[b] (Mrows x K f32) * Bm[b] (K x 65536 f32)
// Block 256 thr = 8 waves; tile 64(M) x 64(N); A panel resident in LDS (TDM),
// B streamed 32-K chunks with f32->bf16 transpose staging.
// ---------------------------------------------------------------------------
__global__ __launch_bounds__(256)
void msa_gemm_bf16(const float* __restrict__ A, int lda, long strideA,
                   const float* __restrict__ Bm, long strideB,
                   float* __restrict__ C, long strideC, int K) {
    const int tid  = threadIdx.x;
    const int wave = tid >> 5;
    const int lane = tid & 31;
    const int mw   = wave & 3;      // 4 m-subtiles of 16
    const int nw   = wave >> 2;     // 2 n-groups of 32
    const int n0   = blockIdx.x * 64;
    const int m0   = blockIdx.y * 64;
    const long b   = blockIdx.z;

    A  += b * strideA + (long)m0 * lda;
    Bm += b * strideB;
    C  += b * strideC + (long)m0 * HW + n0;

    __shared__ __attribute__((aligned(128))) float  AsF[64 * 192]; // 48KB f32 A panel
    __shared__ __attribute__((aligned(32)))  __bf16 Bs[64 * 48];   // [n][k] bf16, pad 48

    // ---- load A panel (64 x K) into LDS once ----
#if MSA_HAVE_TDM
    if (tid < 32) {
        msa_tdm_load_2d(A, (unsigned)(uintptr_t)&AsF[0], 64, K, lda);
        __builtin_amdgcn_s_wait_tensorcnt(0);
    }
#else
    for (int i = tid * 4; i < 64 * K; i += 256 * 4) {
        const int row = i / K, k = i % K;
        *(float4*)&AsF[row * K + k] = *(const float4*)(A + (long)row * lda + k);
    }
#endif

    v8f acc[2] = {};
    const int ml = lane & 15;   // fragment row (A) / col (B,C)
    const int kh = lane >> 4;   // K-half select

    for (int k0 = 0; k0 < K; k0 += 32) {
        // stage B chunk: 32 k x 64 n, transposed bf16 into Bs[n][k]
        {
            const int kk = tid >> 3;            // 0..31
            const int nn = (tid & 7) * 8;       // 0,8,..,56
            const float* src = Bm + (long)(k0 + kk) * HW + n0 + nn;
            if (k0 + 32 < K) __builtin_prefetch(src + 32 * HW, 0, 1);
            const float4 f0 = *(const float4*)(src);
            const float4 f1 = *(const float4*)(src + 4);
            Bs[(nn + 0) * 48 + kk] = (__bf16)f0.x;
            Bs[(nn + 1) * 48 + kk] = (__bf16)f0.y;
            Bs[(nn + 2) * 48 + kk] = (__bf16)f0.z;
            Bs[(nn + 3) * 48 + kk] = (__bf16)f0.w;
            Bs[(nn + 4) * 48 + kk] = (__bf16)f1.x;
            Bs[(nn + 5) * 48 + kk] = (__bf16)f1.y;
            Bs[(nn + 6) * 48 + kk] = (__bf16)f1.z;
            Bs[(nn + 7) * 48 + kk] = (__bf16)f1.w;
        }
        __syncthreads();
        // A fragment from f32 LDS panel: element i -> k = kh*16 + i
        const float* ar = &AsF[(mw * 16 + ml) * K + k0 + kh * 16];
        F16u fa;
        fa.q[0] = *(const float4*)(ar);
        fa.q[1] = *(const float4*)(ar + 4);
        fa.q[2] = *(const float4*)(ar + 8);
        fa.q[3] = *(const float4*)(ar + 12);
        v16bf a;
#pragma unroll
        for (int j = 0; j < 16; ++j) a[j] = (__bf16)fa.f[j];
        // two B fragments (n-subtiles), element i -> k = kh*16 + i (aligned 32B run)
#pragma unroll
        for (int st = 0; st < 2; ++st) {
            const v16bf bfr =
                *(const v16bf*)(&Bs[(nw * 32 + st * 16 + ml) * 48 + kh * 16]);
            acc[st] = __builtin_amdgcn_wmma_f32_16x16x32_bf16(
                false, a, false, bfr, (short)0, acc[st], false, false);
        }
        __syncthreads();
    }
#pragma unroll
    for (int st = 0; st < 2; ++st)
#pragma unroll
        for (int r = 0; r < 8; ++r) {
            const int row = mw * 16 + kh * 8 + r;
            C[(long)row * HW + nw * 32 + st * 16 + ml] = acc[st][r];
        }
}

// ---------------------------------------------------------------------------
// K2: depthwise 3x3 SAME conv + per-channel sum-of-squares (for q,k norms)
// ---------------------------------------------------------------------------
__global__ __launch_bounds__(256)
void msa_dwconv_kernel(const float* __restrict__ qkv, const float* __restrict__ wdw,
                       float* __restrict__ out, float* __restrict__ sq) {
    const int w = threadIdx.x;
    const int h = blockIdx.x;
    const int c = blockIdx.y;
    const int b = blockIdx.z;
    const float* in = qkv + ((long)b * 576 + c) * HW;
    const float* wt = wdw + c * 9;
    float acc = 0.f;
#pragma unroll
    for (int dy = -1; dy <= 1; ++dy) {
        const int y = h + dy;
        if ((unsigned)y >= 256u) continue;
#pragma unroll
        for (int dx = -1; dx <= 1; ++dx) {
            const int x = w + dx;
            if ((unsigned)x >= 256u) continue;
            acc += in[y * 256 + x] * wt[(dy + 1) * 3 + (dx + 1)];
        }
    }
    out[((long)b * 576 + c) * HW + h * 256 + w] = acc;
    if (c < 2 * DIM) {
        float s = acc * acc;
#pragma unroll
        for (int off = 16; off; off >>= 1) s += __shfl_down(s, off, 32);
        __shared__ float red[8];
        if ((threadIdx.x & 31) == 0) red[threadIdx.x >> 5] = s;
        __syncthreads();
        if (threadIdx.x == 0) {
            float t = 0.f;
#pragma unroll
            for (int i = 0; i < 8; ++i) t += red[i];
            atomicAdd(&sq[(c >= DIM ? 384 : 0) + b * DIM + (c % DIM)], t);
        }
    }
}

// ---------------------------------------------------------------------------
// K3: masked channel Gram  G[bh][c][d] += sum_n q[c,n]*k[d,n]*mask[n]^2 (WMMA)
// ---------------------------------------------------------------------------
__global__ __launch_bounds__(256)
void msa_gram_kernel(const float* __restrict__ qkd, const float* __restrict__ mask,
                     float* __restrict__ G) {
    const int slice = blockIdx.x;
    const int bh    = blockIdx.y;
    const int b     = bh >> 2, hd = bh & 3;
    const float* qp = qkd + ((long)b * 576 + hd * CH) * HW;
    const float* kp = qp + (long)DIM * HW;
    const float* mp = mask + (long)b * HW;
    const int wave = threadIdx.x >> 5, lane = threadIdx.x & 31;
    const int ml = lane & 15, kh = lane >> 4;

    v8f acc[3][3] = {};
    const int base = slice * 1024 + wave * 128;

    for (int kc = 0; kc < 4; ++kc) {
        const int n0 = base + kc * 32;
        const int na = n0 + kh * 16;
        F16u ma;
        ma.q[0] = *(const float4*)(mp + na);
        ma.q[1] = *(const float4*)(mp + na + 4);
        ma.q[2] = *(const float4*)(mp + na + 8);
        ma.q[3] = *(const float4*)(mp + na + 12);
        v16bf a[3];
#pragma unroll
        for (int t = 0; t < 3; ++t) {
            const float* r = qp + (long)(t * 16 + ml) * HW + na;
            F16u fa;
            fa.q[0] = *(const float4*)(r);
            fa.q[1] = *(const float4*)(r + 4);
            fa.q[2] = *(const float4*)(r + 8);
            fa.q[3] = *(const float4*)(r + 12);
            v16bf v;
#pragma unroll
            for (int j = 0; j < 16; ++j) v[j] = (__bf16)(fa.f[j] * ma.f[j]);
            a[t] = v;
        }
        const int nb = n0 + kh * 16;
        F16u mb;
        mb.q[0] = *(const float4*)(mp + nb);
        mb.q[1] = *(const float4*)(mp + nb + 4);
        mb.q[2] = *(const float4*)(mp + nb + 8);
        mb.q[3] = *(const float4*)(mp + nb + 12);
        v16bf bb[3];
#pragma unroll
        for (int t = 0; t < 3; ++t) {
            const float* r = kp + (long)(t * 16 + ml) * HW + nb;
            F16u fb;
            fb.q[0] = *(const float4*)(r);
            fb.q[1] = *(const float4*)(r + 4);
            fb.q[2] = *(const float4*)(r + 8);
            fb.q[3] = *(const float4*)(r + 12);
            v16bf v;
#pragma unroll
            for (int j = 0; j < 16; ++j) v[j] = (__bf16)(fb.f[j] * mb.f[j]);
            bb[t] = v;
        }
#pragma unroll
        for (int mt = 0; mt < 3; ++mt)
#pragma unroll
            for (int nt = 0; nt < 3; ++nt)
                acc[mt][nt] = __builtin_amdgcn_wmma_f32_16x16x32_bf16(
                    false, a[mt], false, bb[nt], (short)0, acc[mt][nt], false, false);
    }

    __shared__ float Gs[9 * 256];
    for (int i = threadIdx.x; i < 9 * 256; i += 256) Gs[i] = 0.f;
    __syncthreads();
#pragma unroll
    for (int mt = 0; mt < 3; ++mt)
#pragma unroll
        for (int nt = 0; nt < 3; ++nt)
#pragma unroll
            for (int r = 0; r < 8; ++r)
                atomicAdd(&Gs[(mt * 3 + nt) * 256 + (kh * 8 + r) * 16 + ml],
                          acc[mt][nt][r]);
    __syncthreads();
    float* Gout = G + (long)bh * CH * CH;
    for (int i = threadIdx.x; i < CH * CH; i += 256) {
        const int c = i / CH, d = i % CH;
        const int t = (c / 16) * 3 + (d / 16);
        atomicAdd(&Gout[i], Gs[t * 256 + (c % 16) * 16 + (d % 16)]);
    }
}

// ---------------------------------------------------------------------------
// K4: scale by temperature/(||q_c||*||k_d||) then row softmax -> attn
// ---------------------------------------------------------------------------
__global__ __launch_bounds__(64)
void msa_softmax_kernel(const float* __restrict__ G, const float* __restrict__ sq,
                        const float* __restrict__ temp, float* __restrict__ attn) {
    const int c  = blockIdx.x;
    const int bh = blockIdx.y;
    const int b  = bh >> 2, hd = bh & 3;
    const int d  = threadIdx.x;
    __shared__ float buf[64];
    __shared__ float red2[2];
    const float rq = fmaxf(sqrtf(sq[b * DIM + hd * CH + c]), 1e-12f);
    float val = -1e30f;
    if (d < CH) {
        const float rk = fmaxf(sqrtf(sq[384 + b * DIM + hd * CH + d]), 1e-12f);
        val = G[((long)bh * CH + c) * CH + d] * temp[hd] / (rq * rk);
    }
    buf[d] = val;
    __syncthreads();
    if (d == 0) {
        float mx = buf[0];
        for (int i = 1; i < CH; ++i) mx = fmaxf(mx, buf[i]);
        red2[0] = mx;
    }
    __syncthreads();
    const float e = (d < CH) ? __expf(val - red2[0]) : 0.f;
    buf[d] = e;
    __syncthreads();
    if (d == 0) {
        float s = 0.f;
        for (int i = 0; i < CH; ++i) s += buf[i];
        red2[1] = s;
    }
    __syncthreads();
    if (d < CH) attn[((long)bh * CH + c) * CH + d] = e / red2[1];
}

// ---------------------------------------------------------------------------
// K4b: fold attn into projection:  M[b] = Wproj * blockdiag(attn[b])
// ---------------------------------------------------------------------------
__global__ __launch_bounds__(256)
void msa_fuse_kernel(const float* __restrict__ wproj, const float* __restrict__ attn,
                     float* __restrict__ M) {
    const int idx = blockIdx.x * 256 + threadIdx.x;  // < 192*192
    const int b   = blockIdx.y;
    const int o = idx / DIM, j = idx % DIM;
    const int hd = j / CH, d = j % CH;
    const float* ap = attn + (long)(b * HEADS + hd) * CH * CH + d;
    const float* wp = wproj + o * DIM + hd * CH;
    float s = 0.f;
#pragma unroll 8
    for (int cc = 0; cc < CH; ++cc) s += wp[cc] * ap[cc * CH];
    M[((long)b * DIM + o) * DIM + j] = s;
}

// ---------------------------------------------------------------------------
extern "C" void kernel_launch(void* const* d_in, const int* in_sizes, int n_in,
                              void* d_out, int out_size, void* d_ws, size_t ws_size,
                              hipStream_t stream) {
    const float* x     = (const float*)d_in[0];
    const float* mask  = (const float*)d_in[1];
    const float* wqkv  = (const float*)d_in[2];
    const float* wdw   = (const float*)d_in[3];
    const float* wproj = (const float*)d_in[4];
    const float* temp  = (const float*)d_in[5];
    float* out = (float*)d_out;

    float* ws   = (float*)d_ws;
    float* qkv  = ws;                       // 2*576*65536
    float* qkvd = qkv  + 2L * 576 * HW;     // 2*576*65536
    float* sq   = qkvd + 2L * 576 * HW;     // 768 (sumsq q | sumsq k)
    float* G    = sq   + 768;               // 2*4*48*48
    float* attn = G    + 2L * HEADS * CH * CH;
    float* Mm   = attn + 2L * HEADS * CH * CH;  // 2*192*192

    const int nz = 768 + 2 * HEADS * CH * CH;
    msa_zero_kernel<<<(nz + 255) / 256, 256, 0, stream>>>(sq, nz);

    // K1: qkv = Wqkv(576x192) @ x(192x65536) per batch
    msa_gemm_bf16<<<dim3(HW / 64, 576 / 64, 2), 256, 0, stream>>>(
        wqkv, DIM, 0L, x, (long)DIM * HW, qkv, 576L * HW, DIM);

    // K2: depthwise 3x3 + sumsq(q,k)
    msa_dwconv_kernel<<<dim3(256, 576, 2), 256, 0, stream>>>(qkv, wdw, qkvd, sq);

    // K3: masked Gram (q' vs k')
    msa_gram_kernel<<<dim3(64, 8), 256, 0, stream>>>(qkvd, mask, G);

    // K4: scale + softmax -> attn
    msa_softmax_kernel<<<dim3(CH, 8), 64, 0, stream>>>(G, sq, temp, attn);

    // K4b: M = Wproj * blockdiag(attn)
    msa_fuse_kernel<<<dim3((DIM * DIM) / 256, 2), 256, 0, stream>>>(wproj, attn, Mm);

    // K5: out = M(192x192) @ v(192x65536) per batch (v = qkvd rows 384..575)
    msa_gemm_bf16<<<dim3(HW / 64, DIM / 64, 2), 256, 0, stream>>>(
        Mm, DIM, (long)DIM * DIM, qkvd + 2L * DIM * HW, 576L * HW,
        out, (long)DIM * HW, DIM);
}
